// SynchronGRU_47150150975949
// MI455X (gfx1250) — matrix operand order repaired and checked by reading, not compiled, not run on hardware
//
#include <hip/hip_runtime.h>

// ---------------------------------------------------------------------------
// Stacked GRU (Keras reset_after) for MI455X / gfx1250.
// B=512, T=256, F=8, U=128, G=3U=384, L=8.
//
//  * 32 blocks x 256 threads (8 waves); block owns 16 batch rows for all
//    layers/timesteps (recurrence is per-row -> no inter-block sync).
//  * Wave w owns gate columns [16w,16w+16) of z | r | h-hat; gate math and the
//    h update run fully in registers (h fp32 lives in 8 regs/thread).
//  * All weight B-fragments are preloaded into registers once per layer
//    (weights are timestep-invariant) -> steady state per wave per step:
//    16x ds_load_b128 (8 A frags) + 24x v_wmma_f32_16x16x32_bf16 + elementwise.
//  * Next timestep's x row is register-prefetched during the WMMA phase.
//  * Gate activations use native v_rcp_f32 (no IEEE division chains).
// ---------------------------------------------------------------------------

typedef __attribute__((ext_vector_type(16))) __bf16 v16bf;
typedef __attribute__((ext_vector_type(8)))  __bf16 v8bf;
typedef __attribute__((ext_vector_type(8)))  float  v8f;

#define B_   512
#define T_   256
#define F_   8
#define U_   128
#define G_   384      // 3*U
#define L_   8
#define BT_  16       // batch rows per block
#define NT_  24       // 384/16 column tiles
#define WX0_ELEMS (NT_ * 1 * 32 * 16)   // layer-0 kernel, K padded to 32
#define WK_ELEMS  (NT_ * 4 * 32 * 16)   // K=128 matrices

static __device__ __forceinline__ v8f wmma_bf16(v16bf a, v16bf b, v8f c) {
  return __builtin_amdgcn_wmma_f32_16x16x32_bf16(
      false, a, false, b, (short)0, c, false, false);
}

// A fragment (16x32 bf16) from row-major LDS [16][stride].
// ISA 16-bit A layout: lane<16 -> M=lane, K = kc*32 + {0..7,16..23};
// lane>=16 -> M=lane-16, K = kc*32 + {8..15,24..31}.
static __device__ __forceinline__ v16bf load_a(const __bf16* base, int stride,
                                               int lane, int kc) {
  const int row = lane & 15;
  const int hs  = lane >> 4;
  const __bf16* p = base + row * stride + kc * 32 + hs * 8;
  union { v16bf v; v8bf h[2]; } u;
  u.h[0] = *(const v8bf*)(p);
  u.h[1] = *(const v8bf*)(p + 16);
  return u.v;
}

// B fragment from pre-swizzled weights: [(ntile*nkc + kc)*32 + lane][16].
static __device__ __forceinline__ v16bf load_b(const __bf16* w, int nkc,
                                               int ntile, int kc, int lane) {
  return *(const v16bf*)(w + (size_t)(((ntile * nkc) + kc) * 32 + lane) * 16);
}

// Branch-free fast activations: v_exp_f32 + v_rcp_f32, no IEEE div chains.
static __device__ __forceinline__ float fsigmoid(float x) {
  return __builtin_amdgcn_rcpf(1.0f + __expf(-x));
}
static __device__ __forceinline__ float ftanh(float x) {
  const float e = __expf(2.0f * x);               // tanh = (e-1)/(e+1)
  return (e - 1.0f) * __builtin_amdgcn_rcpf(e + 1.0f);
}

// ---------------------------------------------------------------------------
// Weight swizzle: src (K x 384 fp32) -> WMMA B-fragment bf16 order.
// B layout (ISA 16-bit B): lane holds column n = ntile*16 + (lane&15); its 16
// values are K = kc*32 + (lane>>4)*8 + {0..7,16..23} (matches A).
// ---------------------------------------------------------------------------
__global__ __launch_bounds__(256) void gru_prep_weights(const float* __restrict__ src,
                                                        int K, int nkc,
                                                        __bf16* __restrict__ dst) {
  const int idx = blockIdx.x * blockDim.x + threadIdx.x;
  const int total = NT_ * nkc * 32;
  if (idx >= total) return;
  const int lane  = idx & 31;
  const int kc    = (idx >> 5) % nkc;
  const int ntile = idx / (32 * nkc);
  const int col   = ntile * 16 + (lane & 15);
  const int kbase = kc * 32 + (lane >> 4) * 8;
  __bf16* d = dst + (size_t)idx * 16;
#pragma unroll
  for (int j = 0; j < 16; ++j) {
    const int k = kbase + j + (j >= 8 ? 8 : 0);
    d[j] = (k < K) ? (__bf16)src[(size_t)k * G_ + col] : (__bf16)0.0f;
  }
}

// ---------------------------------------------------------------------------
// One GRU layer over all timesteps for this block's 16 rows.
// NKX = x-projection K chunks (1 for layer 0 [F=8 padded to 32], else 4).
// ---------------------------------------------------------------------------
template <int NKX>
static __device__ __forceinline__ void gru_layer(
    const float* __restrict__ in,     // (B,T,Fin) fp32
    float*       __restrict__ out,    // (B,T,U) fp32
    const float* __restrict__ hsrc,   // init_h for this layer, offset to rbase
    float*       __restrict__ stout,  // states for this layer, offset to rbase
    const float* __restrict__ b0, const float* __restrict__ b1,
    const __bf16* __restrict__ wx, const __bf16* __restrict__ wh,
    __bf16* hbf, __bf16* xbf,
    int rbase, int tid, int lane, int wv) {
  const int hs   = lane >> 4;
  const int col0 = wv * 16 + (lane & 15);
  const int xs   = (NKX == 1) ? 32 : U_;          // x tile LDS stride

  // gate biases (input + recurrent fold where they always add)
  const float bz  = b0[col0]       + b1[col0];
  const float br  = b0[128 + col0] + b1[128 + col0];
  const float bxh = b0[256 + col0];
  const float bhh = b1[256 + col0];

  // ---- preload ALL weight B-fragments into registers (timestep-invariant) --
  v16bf whz[4], whr[4], whh[4];
#pragma unroll
  for (int kc = 0; kc < 4; ++kc) {
    whz[kc] = load_b(wh, 4, wv,      kc, lane);
    whr[kc] = load_b(wh, 4, 8 + wv,  kc, lane);
    whh[kc] = load_b(wh, 4, 16 + wv, kc, lane);
  }
  v16bf wxz[NKX], wxr[NKX], wxh[NKX];
#pragma unroll
  for (int kc = 0; kc < NKX; ++kc) {
    wxz[kc] = load_b(wx, NKX, wv,      kc, lane);
    wxr[kc] = load_b(wx, NKX, 8 + wv,  kc, lane);
    wxh[kc] = load_b(wx, NKX, 16 + wv, kc, lane);
  }

  // ---- h: fp32 in registers (this thread owns rows v+8*hs at column col0) --
  float hreg[8];
#pragma unroll
  for (int v = 0; v < 8; ++v)
    hreg[v] = hsrc[(size_t)(v + hs * 8) * U_ + col0];
  for (int i = tid; i < BT_ * U_; i += 256)       // shared bf16 copy of h
    hbf[i] = (__bf16)hsrc[i];

  // ---- x staging setup + register prefetch of t=0 ----
  float xr[NKX == 1 ? 2 : 8];
  const float* px0;
  const float* px1 = nullptr;
  int   xoff0, xoff1 = 0;
  bool  xvalid = true;
  if constexpr (NKX == 1) {                       // layer 0: 16x32 (F=8 padded)
    const int c = tid & 31;
    xvalid = (c < F_);
    const int r0 = tid >> 5, r1 = r0 + 8;
    px0 = in + (size_t)(rbase + r0) * T_ * F_ + c;
    px1 = in + (size_t)(rbase + r1) * T_ * F_ + c;
    xoff0 = r0 * 32 + c;
    xoff1 = r1 * 32 + c;
    xr[0] = xvalid ? px0[0] : 0.0f;
    xr[1] = xvalid ? px1[0] : 0.0f;
  } else {                                        // 16x128, 8 floats/thread
    const int row = tid >> 4;
    const int c0  = (tid & 15) * 8;
    px0   = in + (size_t)(rbase + row) * T_ * U_ + c0;
    xoff0 = row * U_ + c0;
    const float4 a = ((const float4*)px0)[0];
    const float4 b = ((const float4*)px0)[1];
    xr[0]=a.x; xr[1]=a.y; xr[2]=a.z; xr[3]=a.w;
    xr[4]=b.x; xr[5]=b.y; xr[6]=b.z; xr[7]=b.w;
  }

  // output base: rows hs*8 + v, column col0
  float* outp = out + (size_t)(rbase + hs * 8) * T_ * U_ + col0;

  __syncthreads();                                // hbf (and prior layer) ready

  for (int t = 0; t < T_; ++t) {
    // ---- commit staged x registers to LDS (bf16) ----
    if constexpr (NKX == 1) {
      xbf[xoff0] = (__bf16)xr[0];
      xbf[xoff1] = (__bf16)xr[1];
    } else {
#pragma unroll
      for (int j = 0; j < 8; ++j) xbf[xoff0 + j] = (__bf16)xr[j];
    }
    __syncthreads();                              // x ready; h writes published

    // ---- register-prefetch next timestep's x (hidden behind WMMAs) ----
    if (t + 1 < T_) {
      if constexpr (NKX == 1) {
        xr[0] = xvalid ? px0[(size_t)(t + 1) * F_] : 0.0f;
        xr[1] = xvalid ? px1[(size_t)(t + 1) * F_] : 0.0f;
      } else {
        const float* p = px0 + (size_t)(t + 1) * U_;
        const float4 a = ((const float4*)p)[0];
        const float4 b = ((const float4*)p)[1];
        xr[0]=a.x; xr[1]=a.y; xr[2]=a.z; xr[3]=a.w;
        xr[4]=b.x; xr[5]=b.y; xr[6]=b.z; xr[7]=b.w;
        if (t + 2 < T_)
          __builtin_prefetch(px0 + (size_t)(t + 2) * U_, 0, 3);
      }
    }

    // ---- GEMMs: B matrices from registers, A fragments from LDS ----
    v8f accz  = {bz, bz, bz, bz, bz, bz, bz, bz};
    v8f accr  = {br, br, br, br, br, br, br, br};
    v8f accxh = {bxh, bxh, bxh, bxh, bxh, bxh, bxh, bxh};
    v8f acchh = {bhh, bhh, bhh, bhh, bhh, bhh, bhh, bhh};
#pragma unroll
    for (int kc = 0; kc < 4; ++kc) {              // h @ rec_kernel
      const v16bf a = load_a(hbf, U_, lane, kc);
      accz  = wmma_bf16(a, whz[kc], accz);
      accr  = wmma_bf16(a, whr[kc], accr);
      acchh = wmma_bf16(a, whh[kc], acchh);
    }
#pragma unroll
    for (int kc = 0; kc < NKX; ++kc) {            // x @ kernel
      const v16bf a = load_a(xbf, xs, lane, kc);
      accz  = wmma_bf16(a, wxz[kc], accz);
      accr  = wmma_bf16(a, wxr[kc], accr);
      accxh = wmma_bf16(a, wxh[kc], accxh);
    }
    __syncthreads();                              // all hbf/xbf reads done

    // ---- gates + state update, fully in registers ----
    float* outt = outp + (size_t)t * U_;
#pragma unroll
    for (int v = 0; v < 8; ++v) {
      const float z  = fsigmoid(accz[v]);
      const float r  = fsigmoid(accr[v]);
      const float hh = ftanh(accxh[v] + r * acchh[v]);
      const float hn = hh + z * (hreg[v] - hh);   // z*h + (1-z)*hh
      hreg[v] = hn;
      hbf[(v + hs * 8) * U_ + col0] = (__bf16)hn; // feeds next step's A frags
      outt[(size_t)v * T_ * U_] = hn;             // out[rbase+row][t][col0]
    }
    // next iteration's post-x barrier publishes the hbf writes
  }

  // final hidden state -> states[l]
#pragma unroll
  for (int v = 0; v < 8; ++v)
    stout[(size_t)(v + hs * 8) * U_ + col0] = hreg[v];
  __syncthreads();                                // before next layer's h init
}

// ---------------------------------------------------------------------------
__global__ __launch_bounds__(256) void gru_main(const float* __restrict__ xin,
                                                const float* __restrict__ inith,
                                                const float* __restrict__ biases,
                                                const __bf16* __restrict__ wz,
                                                float* __restrict__ ping,    // d_ws (B,T,U)
                                                float* __restrict__ outb) {  // d_out
  __shared__ __align__(32) __bf16 hbf[BT_ * U_];
  __shared__ __align__(32) __bf16 xbf[BT_ * U_];

  const int tid   = threadIdx.x;
  const int lane  = tid & 31;
  const int wv    = tid >> 5;
  const int rbase = blockIdx.x * BT_;
  float* states = outb + (size_t)B_ * T_ * U_;

  const __bf16* whbase = wz + WX0_ELEMS + (size_t)7 * WK_ELEMS;

  // layer 0: F=8 (padded to one K-chunk)
  gru_layer<1>(xin, ping,
               inith + (size_t)rbase * U_,
               states + (size_t)rbase * U_,
               biases, biases + G_,
               wz, whbase,
               hbf, xbf, rbase, tid, lane, wv);

  // layers 1..7: K=128
  for (int l = 1; l < L_; ++l) {
    const float* in  = (l & 1) ? ping : outb;
    float*       out = (l & 1) ? outb : ping;     // l=7 -> outb (d_out)
    gru_layer<4>(in, out,
                 inith + ((size_t)l * B_ + rbase) * U_,
                 states + ((size_t)l * B_ + rbase) * U_,
                 biases + (size_t)l * 2 * G_, biases + (size_t)l * 2 * G_ + G_,
                 wz + WX0_ELEMS + (size_t)(l - 1) * WK_ELEMS,
                 whbase + (size_t)l * WK_ELEMS,
                 hbf, xbf, rbase, tid, lane, wv);
  }
}

// ---------------------------------------------------------------------------
// d_in: 0=input(B,T,F) 1=init_h(L,B,U) 2=kernel0(F,3U) 3=kernels(L-1,U,3U)
//       4=rec_kernels(L,U,3U) 5=biases(L,2,3U)
// d_ws: [0, B*T*U*4) fp32 ping buffer, then swizzled bf16 weights (~1.5 MB).
// ---------------------------------------------------------------------------
extern "C" void kernel_launch(void* const* d_in, const int* in_sizes, int n_in,
                              void* d_out, int out_size, void* d_ws, size_t ws_size,
                              hipStream_t stream) {
  (void)in_sizes; (void)n_in; (void)out_size; (void)ws_size;
  const float* xin    = (const float*)d_in[0];
  const float* inith  = (const float*)d_in[1];
  const float* k0     = (const float*)d_in[2];
  const float* ks     = (const float*)d_in[3];
  const float* rks    = (const float*)d_in[4];
  const float* biases = (const float*)d_in[5];

  float*  ping = (float*)d_ws;
  __bf16* wz   = (__bf16*)((char*)d_ws + (size_t)B_ * T_ * U_ * sizeof(float));

  gru_prep_weights<<<(NT_ * 1 * 32 + 255) / 256, 256, 0, stream>>>(k0, F_, 1, wz);
  for (int l = 1; l < L_; ++l)
    gru_prep_weights<<<(NT_ * 4 * 32 + 255) / 256, 256, 0, stream>>>(
        ks + (size_t)(l - 1) * U_ * G_, U_, 4,
        wz + WX0_ELEMS + (size_t)(l - 1) * WK_ELEMS);
  for (int l = 0; l < L_; ++l)
    gru_prep_weights<<<(NT_ * 4 * 32 + 255) / 256, 256, 0, stream>>>(
        rks + (size_t)l * U_ * G_, U_, 4,
        wz + WX0_ELEMS + (size_t)7 * WK_ELEMS + (size_t)l * WK_ELEMS);

  gru_main<<<B_ / BT_, 256, 0, stream>>>(xin, inith, biases, wz, ping, (float*)d_out);
}
